// SwinTransformer_87454124081371
// MI455X (gfx1250) — compile-verified, matrix operand order checked by reading, not run on hardware
//
#include <hip/hip_runtime.h>
#include <hip/hip_bf16.h>
#include <math.h>
#include <stdint.h>

// ---------------- problem constants ----------------
#define BB   32
#define HH   56
#define WW_  56
#define CC   384
#define WS   7
#define SHIFT 3
#define HEADS 12
#define HD   32                 // CC / HEADS
#define MLP  1536
#define NTOK 49                 // WS*WS
#define NWIN 64                 // (H/WS)*(W/WS)
#define MTOK (BB * HH * WW_)    // 100352 rows
#define QKVC (3 * CC)           // 1152
#define SCALE 0.17677669529663687f  // 1/sqrt(32)

typedef __attribute__((ext_vector_type(16))) _Float16 v16h;
typedef __attribute__((ext_vector_type(8)))  _Float16 v8h;
typedef __attribute__((ext_vector_type(8)))  float    v8f;

// ---------------------------------------------------------------------------
// CDNA5 async global->LDS copy (ASYNCcnt-tracked, bypasses VGPRs).
// VDST VGPR holds the LDS byte address (low 32 bits of the flat LDS pointer),
// VADDR pair holds the 64-bit global address ("GV" mode, saddr = off).
// ---------------------------------------------------------------------------
__device__ __forceinline__ void async_ld_b128(const void* gptr, void* lptr) {
    uint32_t lds = (uint32_t)(uintptr_t)lptr;          // LDS aperture: addr[31:0]
    uint64_t ga  = (uint64_t)(uintptr_t)gptr;
    asm volatile("global_load_async_to_lds_b128 %0, %1, off"
                 :: "v"(lds), "v"(ga) : "memory");
}

#if __has_builtin(__builtin_amdgcn_s_wait_asynccnt)
#define WAIT_ASYNC(n) __builtin_amdgcn_s_wait_asynccnt(n)
#else
#define WAIT_ASYNC(n) asm volatile("s_wait_asynccnt %0" :: "i"(n) : "memory")
#endif

// ---------------------------------------------------------------------------
// 16x32 f16 fragment loader (A-matrix layout, also used for B^T tiles).
// lanes 0-15: M=lane, K={0..7,16..23}; lanes 16-31: M=lane-16, K={8..15,24..31}
// ---------------------------------------------------------------------------
__device__ __forceinline__ v16h load_frag(const _Float16* p, int ld) {
    int lane = threadIdx.x & 31;
    const _Float16* q = p + (size_t)(lane & 15) * ld + ((lane >> 4) << 3);
    v8h lo = *(const v8h*)(q);
    v8h hi = *(const v8h*)(q + 16);
    return __builtin_shufflevector(lo, hi, 0,1,2,3,4,5,6,7,8,9,10,11,12,13,14,15);
}

__device__ __forceinline__ v8f wmma_f16(v16h a, v16h b, v8f c) {
    return __builtin_amdgcn_wmma_f32_16x16x32_f16(false, a, false, b, (short)0, c,
                                                  false, false);
}

__device__ __forceinline__ float gelu_exact(float x) {
    return 0.5f * x * (1.0f + erff(x * 0.70710678118654752f));
}

// ---------------------------------------------------------------------------
// Weight transpose + f32 -> f16 convert:  w (K x N) -> wt (N x K)
// ---------------------------------------------------------------------------
__global__ void transpose_w_kernel(const float* __restrict__ w,
                                   _Float16* __restrict__ wt, int K, int N) {
    int idx = blockIdx.x * 256 + threadIdx.x;
    if (idx < K * N) {
        int n = idx / K, k = idx % K;
        wt[idx] = (_Float16)w[(size_t)k * N + n];
    }
}

// ---------------------------------------------------------------------------
// LN1 + roll(-SHIFT,-SHIFT) + window partition, fp32 in -> f16 windowed out
// ---------------------------------------------------------------------------
__global__ __launch_bounds__(384)
void ln1_shift_kernel(const float* __restrict__ x, const float* __restrict__ g,
                      const float* __restrict__ b, _Float16* __restrict__ xw) {
    int row = blockIdx.x;
    int t = threadIdx.x;
    float v = x[(size_t)row * CC + t];
    float s = v, s2 = v * v;
    #pragma unroll
    for (int o = 16; o > 0; o >>= 1) {
        s  += __shfl_down(s,  o, 32);
        s2 += __shfl_down(s2, o, 32);
    }
    __shared__ float ps[12], ps2[12], mb[2];
    if ((t & 31) == 0) { ps[t >> 5] = s; ps2[t >> 5] = s2; }
    __syncthreads();
    if (t == 0) {
        float a = 0.f, a2 = 0.f;
        #pragma unroll
        for (int i = 0; i < 12; ++i) { a += ps[i]; a2 += ps2[i]; }
        float mean = a * (1.0f / CC);
        float var  = a2 * (1.0f / CC) - mean * mean;
        mb[0] = mean; mb[1] = rsqrtf(var + 1e-5f);
    }
    __syncthreads();
    float y = (v - mb[0]) * mb[1] * g[t] + b[t];
    int bb = row / (HH * WW_), rem = row % (HH * WW_);
    int h = rem / WW_, w = rem % WW_;
    int hp = (h + HH - SHIFT) % HH;
    int wp = (w + WW_ - SHIFT) % WW_;
    int drow = (bb * NWIN + (hp / WS) * (WW_ / WS) + (wp / WS)) * NTOK
             + (hp % WS) * WS + (wp % WS);
    xw[(size_t)drow * CC + t] = (_Float16)y;
}

// ---------------------------------------------------------------------------
// LN2: fp32 x1 -> f16, natural layout
// ---------------------------------------------------------------------------
__global__ __launch_bounds__(384)
void ln2_kernel(const float* __restrict__ x, const float* __restrict__ g,
                const float* __restrict__ b, _Float16* __restrict__ y) {
    int row = blockIdx.x;
    int t = threadIdx.x;
    float v = x[(size_t)row * CC + t];
    float s = v, s2 = v * v;
    #pragma unroll
    for (int o = 16; o > 0; o >>= 1) {
        s  += __shfl_down(s,  o, 32);
        s2 += __shfl_down(s2, o, 32);
    }
    __shared__ float ps[12], ps2[12], mb[2];
    if ((t & 31) == 0) { ps[t >> 5] = s; ps2[t >> 5] = s2; }
    __syncthreads();
    if (t == 0) {
        float a = 0.f, a2 = 0.f;
        #pragma unroll
        for (int i = 0; i < 12; ++i) { a += ps[i]; a2 += ps2[i]; }
        float mean = a * (1.0f / CC);
        float var  = a2 * (1.0f / CC) - mean * mean;
        mb[0] = mean; mb[1] = rsqrtf(var + 1e-5f);
    }
    __syncthreads();
    float o = (v - mb[0]) * mb[1] * g[t] + b[t];
    y[(size_t)row * CC + t] = (_Float16)o;
}

// ---------------------------------------------------------------------------
// WMMA GEMM:  out(M x N) = A(M x K, f16) * Bt(N x K, f16)^T + bias
// Block tile 128x128, 256 threads = 8 waves (4 M x 2 N), wave tile 32x64
// (8 wmma per K-step).  Double-buffered LDS staged with async global->LDS;
// software pipeline: issue stage k+1, s_wait_asynccnt 4 (in-order) => stage k
// resident, barrier, 8 WMMAs, barrier.
// EPI: 0 qkv->f16 | 1 proj + window-reverse+roll + residual -> f32
//      2 fc1+GELU->f16 | 3 fc2 + residual -> f32
// ---------------------------------------------------------------------------
#define LDA 40   // 32 halves + 8 pad

template <int EPI>
__global__ __launch_bounds__(256)
void gemm_wmma_kernel(const _Float16* __restrict__ A,
                      const _Float16* __restrict__ Bt,
                      const float* __restrict__ bias,
                      float* __restrict__ outf,
                      _Float16* __restrict__ outh,
                      const float* __restrict__ resid,
                      int M, int N, int K) {
    __shared__ __align__(16) _Float16 As[2][128 * LDA];
    __shared__ __align__(16) _Float16 Bs[2][128 * LDA];

    int tid = threadIdx.x;
    int m0 = blockIdx.y * 128;
    int n0 = blockIdx.x * 128;
    int wave = tid >> 5;
    int lane = tid & 31;
    int wm = wave >> 1;           // 0..3  (M)
    int wn = wave & 1;            // 0..1  (N)

    // staging addresses for this thread: 2 chunks of A + 2 of B per stage
    int r0 = tid >> 2, r1 = (tid + 256) >> 2;
    int off0 = (tid & 3) * 8, off1 = ((tid + 256) & 3) * 8;

    v8f acc[2][4] = {};
    int nk = K / 32;

    // issue stage 0
    {
        async_ld_b128(&A [(size_t)(m0 + r0) * K + off0], &As[0][r0 * LDA + off0]);
        async_ld_b128(&A [(size_t)(m0 + r1) * K + off1], &As[0][r1 * LDA + off1]);
        async_ld_b128(&Bt[(size_t)(n0 + r0) * K + off0], &Bs[0][r0 * LDA + off0]);
        async_ld_b128(&Bt[(size_t)(n0 + r1) * K + off1], &Bs[0][r1 * LDA + off1]);
    }

    for (int kt = 0; kt < nk; ++kt) {
        int cur = kt & 1;
        if (kt + 1 < nk) {
            int nxt = cur ^ 1, k1 = (kt + 1) * 32;
            async_ld_b128(&A [(size_t)(m0 + r0) * K + k1 + off0], &As[nxt][r0 * LDA + off0]);
            async_ld_b128(&A [(size_t)(m0 + r1) * K + k1 + off1], &As[nxt][r1 * LDA + off1]);
            async_ld_b128(&Bt[(size_t)(n0 + r0) * K + k1 + off0], &Bs[nxt][r0 * LDA + off0]);
            async_ld_b128(&Bt[(size_t)(n0 + r1) * K + k1 + off1], &Bs[nxt][r1 * LDA + off1]);
            WAIT_ASYNC(4);
        } else {
            WAIT_ASYNC(0);
        }
        __syncthreads();

        v16h af[2], bf[4];
        #pragma unroll
        for (int i = 0; i < 2; ++i)
            af[i] = load_frag(&As[cur][(wm * 32 + i * 16) * LDA], LDA);
        #pragma unroll
        for (int j = 0; j < 4; ++j)
            bf[j] = load_frag(&Bs[cur][(wn * 64 + j * 16) * LDA], LDA);
        #pragma unroll
        for (int i = 0; i < 2; ++i)
            #pragma unroll
            for (int j = 0; j < 4; ++j)
                acc[i][j] = wmma_f16(af[i], bf[j], acc[i][j]);
        __syncthreads();
    }

    // epilogue: per tile, VGPR r holds (M = r or r+8, N = lane%16)
    int rbase = (lane >> 4) << 3;
    int cloc  = lane & 15;
    #pragma unroll
    for (int i = 0; i < 2; ++i) {
        #pragma unroll
        for (int j = 0; j < 4; ++j) {
            #pragma unroll
            for (int r = 0; r < 8; ++r) {
                int gM = m0 + wm * 32 + i * 16 + rbase + r;
                int gN = n0 + wn * 64 + j * 16 + cloc;
                float val = acc[i][j][r] + bias[gN];
                if (EPI == 0) {
                    outh[(size_t)gM * N + gN] = (_Float16)val;
                } else if (EPI == 1) {
                    int win = gM / NTOK, n = gM % NTOK;
                    int b  = win / NWIN, wi = win % NWIN;
                    int wh = wi / (WW_ / WS), ww = wi % (WW_ / WS);
                    int h  = (wh * WS + n / WS + SHIFT) % HH;
                    int w  = (ww * WS + n % WS + SHIFT) % WW_;
                    size_t idx = ((size_t)b * (HH * WW_) + h * WW_ + w) * CC + gN;
                    outf[idx] = resid[idx] + val;
                } else if (EPI == 2) {
                    outh[(size_t)gM * N + gN] = (_Float16)gelu_exact(val);
                } else {
                    size_t idx = (size_t)gM * N + gN;
                    outf[idx] = resid[idx] + val;
                }
            }
        }
    }
}

// ---------------------------------------------------------------------------
// Windowed attention: one block per (window, head), 128 threads = 4 waves.
// Q/K staged via async global->LDS; V transposed through VGPRs.
// ---------------------------------------------------------------------------
__device__ __forceinline__ int shift_region(int u) {
    return (u < HH - WS) ? 0 : ((u < HH - SHIFT) ? 1 : 2);
}

#define LDP 72   // 64 halves + 8 pad

__global__ __launch_bounds__(128)
void attn_kernel(const _Float16* __restrict__ qkv,
                 const float* __restrict__ rel_bias,
                 _Float16* __restrict__ attn_out) {
    int head = blockIdx.x;
    int win  = blockIdx.y;
    int tid  = threadIdx.x;
    int wave = tid >> 5;
    int lane = tid & 31;

    __shared__ __align__(16) _Float16 Qs[64 * LDA];
    __shared__ __align__(16) _Float16 Ks[64 * LDA];
    __shared__ __align__(16) _Float16 Vt[32 * LDP];   // V^T: [d][m]
    __shared__ __align__(16) float    Sf[64 * 64];
    __shared__ __align__(16) _Float16 Ps[64 * LDP];

    // async-stage Q and K in 16B chunks (4 chunks per 32-half row)
    #pragma unroll
    for (int it = 0; it < 2; ++it) {
        int idx = tid + it * 128;          // 0..255
        int m = idx >> 2, off = (idx & 3) * 8;
        if (m < NTOK) {
            size_t base = (size_t)(win * NTOK + m) * QKVC + head * HD + off;
            async_ld_b128(&qkv[base],      &Qs[m * LDA + off]);
            async_ld_b128(&qkv[base + CC], &Ks[m * LDA + off]);
        } else {
            v8h z = {};
            *(v8h*)&Qs[m * LDA + off] = z;
            *(v8h*)&Ks[m * LDA + off] = z;
        }
    }
    // V transposed into LDS
    for (int idx = tid; idx < 32 * 64; idx += 128) {
        int d = idx >> 6, m = idx & 63;
        Vt[d * LDP + m] = (m < NTOK)
            ? qkv[(size_t)(win * NTOK + m) * QKVC + head * HD + 2 * CC + d]
            : (_Float16)0.f;
    }
    WAIT_ASYNC(0);
    __syncthreads();

    // ---- scores: each wave owns 16 rows x 64 cols (4 wmma tiles) ----
    {
        v8f sacc[4] = {};
        v16h aq = load_frag(&Qs[(wave * 16) * LDA], LDA);
        #pragma unroll
        for (int j = 0; j < 4; ++j) {
            v16h bk = load_frag(&Ks[(j * 16) * LDA], LDA);
            sacc[j] = wmma_f16(aq, bk, sacc[j]);
        }
        int wi = win % NWIN;
        int wh = wi / (WW_ / WS), ww = wi % (WW_ / WS);
        int rbase = (lane >> 4) << 3;
        int cloc  = lane & 15;
        #pragma unroll
        for (int j = 0; j < 4; ++j) {
            #pragma unroll
            for (int r = 0; r < 8; ++r) {
                int row = wave * 16 + rbase + r;
                int col = j * 16 + cloc;
                float val = sacc[j][r] * SCALE;
                if (row < NTOK && col < NTOK) {
                    int r1 = row / WS, c1 = row % WS;
                    int r2 = col / WS, c2 = col % WS;
                    int ridx = (r1 - r2 + WS - 1) * (2 * WS - 1) + (c1 - c2 + WS - 1);
                    val += rel_bias[ridx * HEADS + head];
                    int cn = 3 * shift_region(wh * WS + r1) + shift_region(ww * WS + c1);
                    int cm = 3 * shift_region(wh * WS + r2) + shift_region(ww * WS + c2);
                    if (cn != cm) val -= 100.0f;
                }
                Sf[row * 64 + col] = val;
            }
        }
    }
    __syncthreads();

    // ---- softmax over 49 valid cols ----
    if (tid < 64) {
        int row = tid;
        if (row < NTOK) {
            float mx = -1e30f;
            for (int m = 0; m < NTOK; ++m) mx = fmaxf(mx, Sf[row * 64 + m]);
            float sum = 0.f;
            float e[NTOK];
            for (int m = 0; m < NTOK; ++m) { e[m] = expf(Sf[row * 64 + m] - mx); sum += e[m]; }
            float inv = 1.0f / sum;
            for (int m = 0; m < NTOK; ++m) Ps[row * LDP + m] = (_Float16)(e[m] * inv);
            for (int m = NTOK; m < 64; ++m) Ps[row * LDP + m] = (_Float16)0.f;
        } else {
            for (int m = 0; m < 64; ++m) Ps[row * LDP + m] = (_Float16)0.f;
        }
    }
    __syncthreads();

    // ---- O = P @ V : each wave 16 rows x 32 cols, K = 64 (2 steps) ----
    {
        v8f oacc[2] = {};
        #pragma unroll
        for (int kk = 0; kk < 2; ++kk) {
            v16h ap = load_frag(&Ps[(wave * 16) * LDP + kk * 32], LDP);
            #pragma unroll
            for (int j = 0; j < 2; ++j) {
                v16h bv = load_frag(&Vt[(j * 16) * LDP + kk * 32], LDP);
                oacc[j] = wmma_f16(ap, bv, oacc[j]);
            }
        }
        int rbase = (lane >> 4) << 3;
        int cloc  = lane & 15;
        #pragma unroll
        for (int j = 0; j < 2; ++j) {
            #pragma unroll
            for (int r = 0; r < 8; ++r) {
                int row = wave * 16 + rbase + r;
                int col = j * 16 + cloc;
                if (row < NTOK)
                    attn_out[(size_t)(win * NTOK + row) * CC + head * HD + col] =
                        (_Float16)oacc[j][r];
            }
        }
    }
}

// ---------------------------------------------------------------------------
// launch
// ---------------------------------------------------------------------------
extern "C" void kernel_launch(void* const* d_in, const int* in_sizes, int n_in,
                              void* d_out, int out_size, void* d_ws, size_t ws_size,
                              hipStream_t stream) {
    const float* x      = (const float*)d_in[0];
    const float* ln1_g  = (const float*)d_in[1];
    const float* ln1_b  = (const float*)d_in[2];
    const float* qkv_w  = (const float*)d_in[3];
    const float* qkv_b  = (const float*)d_in[4];
    const float* relb   = (const float*)d_in[5];
    const float* proj_w = (const float*)d_in[6];
    const float* proj_b = (const float*)d_in[7];
    const float* ln2_g  = (const float*)d_in[8];
    const float* ln2_b  = (const float*)d_in[9];
    const float* mlp_w1 = (const float*)d_in[10];
    const float* mlp_b1 = (const float*)d_in[11];
    const float* mlp_w2 = (const float*)d_in[12];
    const float* mlp_b2 = (const float*)d_in[13];
    float* out = (float*)d_out;

    char* ws = (char*)d_ws;
    const size_t XW_B   = (size_t)MTOK * CC * 2;
    const size_t QKV_B  = (size_t)MTOK * QKVC * 2;
    const size_t AO_B   = XW_B;
    const size_t X1_B   = (size_t)MTOK * CC * 4;
    _Float16* xw   = (_Float16*)(ws);                    // reused as y2
    _Float16* qkvb = (_Float16*)(ws + XW_B);             // region reused as h1
    _Float16* ao   = (_Float16*)(ws + XW_B + QKV_B);
    float*    x1   = (float*)   (ws + XW_B + QKV_B + AO_B);
    char*     wsw  = ws + XW_B + QKV_B + AO_B + X1_B;
    _Float16* qkvT = (_Float16*)(wsw);
    _Float16* projT= (_Float16*)(wsw + (size_t)QKVC * CC * 2);
    _Float16* w1T  = (_Float16*)(wsw + (size_t)QKVC * CC * 2 + (size_t)CC * CC * 2);
    _Float16* w2T  = (_Float16*)(wsw + (size_t)QKVC * CC * 2 + (size_t)CC * CC * 2
                                      + (size_t)CC * MLP * 2);
    _Float16* y2 = xw;
    _Float16* h1 = qkvb;

    transpose_w_kernel<<<(CC * QKVC + 255) / 256, 256, 0, stream>>>(qkv_w, qkvT, CC, QKVC);
    transpose_w_kernel<<<(CC * CC   + 255) / 256, 256, 0, stream>>>(proj_w, projT, CC, CC);
    transpose_w_kernel<<<(CC * MLP  + 255) / 256, 256, 0, stream>>>(mlp_w1, w1T, CC, MLP);
    transpose_w_kernel<<<(MLP * CC  + 255) / 256, 256, 0, stream>>>(mlp_w2, w2T, MLP, CC);

    ln1_shift_kernel<<<MTOK, 384, 0, stream>>>(x, ln1_g, ln1_b, xw);

    gemm_wmma_kernel<0><<<dim3(QKVC / 128, MTOK / 128), 256, 0, stream>>>(
        xw, qkvT, qkv_b, nullptr, qkvb, nullptr, MTOK, QKVC, CC);

    attn_kernel<<<dim3(HEADS, BB * NWIN), 128, 0, stream>>>(qkvb, relb, ao);

    gemm_wmma_kernel<1><<<dim3(CC / 128, MTOK / 128), 256, 0, stream>>>(
        ao, projT, proj_b, x1, nullptr, x, MTOK, CC, CC);

    ln2_kernel<<<MTOK, 384, 0, stream>>>(x1, ln2_g, ln2_b, y2);

    gemm_wmma_kernel<2><<<dim3(MLP / 128, MTOK / 128), 256, 0, stream>>>(
        y2, w1T, mlp_b1, nullptr, h1, nullptr, MTOK, MLP, CC);

    gemm_wmma_kernel<3><<<dim3(CC / 128, MTOK / 128), 256, 0, stream>>>(
        h1, w2T, mlp_b2, out, nullptr, x1, MTOK, CC, MLP);
}
